// Lift3DClipProgressiveReplacementGraspNet_16733192585546
// MI455X (gfx1250) — matrix operand order repaired and checked by reading, not compiled, not run on hardware
//
#include <hip/hip_runtime.h>

// Problem constants
#define BB    32
#define NS    2048
#define NC    512
#define FD    768
#define CH    256
#define EPSV  1e-5f
#define ALPHA 0.5f

// Tile: 256(M) x 64(N seed columns) per 256-thread workgroup (8 waves).
// Each wave owns 2 m-tiles x 4 n-tiles of 16x16.
#define NT    64

// Padded LDS row lengths (elements)
#define FDP (FD + 16)   // lift tile row, bf16 (row stride 1568B, 32B-mult)
#define XFP (CH + 1)    // x f32 row (bank-conflict-free column reduce)
#define CHP (CH + 16)   // bf16 rows for x_ln / h (row stride 544B, 32B-mult)

typedef __attribute__((ext_vector_type(8)))  float   v8f;
typedef __attribute__((ext_vector_type(8)))  __bf16  v8bf;
typedef __attribute__((ext_vector_type(16))) __bf16  v16bf;

// Workspace layout (bytes)
#define WS_IDX   0
#define WS_WP    (BB * NS * 4)                 // 262144
#define WS_WA1   (WS_WP + CH * FD * 2)         // +393216
#define WS_WA2   (WS_WA1 + CH * CH * 2)        // +131072

// ---------------------------------------------------------------------------
// Kernel 1: nearest-center argmin over 512 centers (centers cached in LDS)
// ---------------------------------------------------------------------------
__global__ void __launch_bounds__(256)
nn_argmin(const float* __restrict__ seed, const float* __restrict__ ctr,
          int* __restrict__ idx)
{
    __shared__ float sc[NC * 3];
    const int b  = blockIdx.y;
    const int si = blockIdx.x * 256 + threadIdx.x;

    for (int i = threadIdx.x; i < NC * 3; i += 256)
        sc[i] = ctr[(size_t)b * NC * 3 + i];
    __syncthreads();

    const size_t sb = ((size_t)b * NS + si) * 3;
    const float sx = seed[sb + 0], sy = seed[sb + 1], sz = seed[sb + 2];

    float best = 3.4e38f;
    int   bi   = 0;
    #pragma unroll 4
    for (int c = 0; c < NC; ++c) {
        const float dx = sx - sc[c * 3 + 0];
        const float dy = sy - sc[c * 3 + 1];
        const float dz = sz - sc[c * 3 + 2];
        const float d2 = dx * dx + dy * dy + dz * dz;
        if (d2 < best) { best = d2; bi = c; }   // first-min, matches argmin
    }
    idx[(size_t)b * NS + si] = bi;
}

// ---------------------------------------------------------------------------
// Kernel 2: convert the three weight matrices to bf16 (once per launch)
// ---------------------------------------------------------------------------
__global__ void __launch_bounds__(256)
cvt_weights(const float* __restrict__ wp, const float* __restrict__ wa1,
            const float* __restrict__ wa2, __bf16* __restrict__ dwp,
            __bf16* __restrict__ dwa1, __bf16* __restrict__ dwa2)
{
    const int i = blockIdx.x * 256 + threadIdx.x;
    if (i < CH * FD) dwp[i] = (__bf16)wp[i];
    if (i < CH * CH) { dwa1[i] = (__bf16)wa1[i]; dwa2[i] = (__bf16)wa2[i]; }
}

// ---------------------------------------------------------------------------
// Fragment loaders following the CDNA5 WMMA VGPR layout tables (wave32).
// A (16x32 bf16): lane m=l%16, hi=l/16; halves 0..7 -> K = k0+hi*8 .. +7,
//                 halves 8..15 -> K = k0+16+hi*8 .. +7  (two 16B loads)
// B (32x16 bf16): lane n=l%16, hi=l/16; same K order as A (two 16B loads)
// ---------------------------------------------------------------------------
__device__ __forceinline__ v16bf load_a_frag(const __bf16* row, int k0, int hi)
{
    v8bf lo = *(const v8bf*)(row + k0 + hi * 8);
    v8bf hh = *(const v8bf*)(row + k0 + 16 + hi * 8);
    return __builtin_shufflevector(lo, hh, 0, 1, 2, 3, 4, 5, 6, 7,
                                           8, 9, 10, 11, 12, 13, 14, 15);
}

__device__ __forceinline__ v16bf load_b_frag(const __bf16* row, int k0, int hi)
{
    v8bf lo = *(const v8bf*)(row + k0 + hi * 8);
    v8bf hh = *(const v8bf*)(row + k0 + 16 + hi * 8);
    return __builtin_shufflevector(lo, hh, 0, 1, 2, 3, 4, 5, 6, 7,
                                           8, 9, 10, 11, 12, 13, 14, 15);
}

// ---------------------------------------------------------------------------
// Kernel 3: fused gather -> proj GEMM -> LayerNorm -> MLP (2 GEMMs) -> blend
// ---------------------------------------------------------------------------
__global__ void __launch_bounds__(256)
fused_lift(const float* __restrict__ feat, const float* __restrict__ fgn,
           const __bf16* __restrict__ Wp,  const float* __restrict__ bp,
           const __bf16* __restrict__ Wa1, const float* __restrict__ ba1,
           const __bf16* __restrict__ Wa2, const float* __restrict__ ba2,
           const int* __restrict__ idx, float* __restrict__ out)
{
    // Region A: lift tile (bf16 NTxFDP = 100352B),
    //           later x f32 (NT*XFP*4 = 65792B) + x_ln bf16 (NT*CHP*2 = 34816B)
    // Region B: LN partials/mu/rs (2560B), later h bf16 (34816B)
    __shared__ __align__(64) unsigned char smA[NT * XFP * 4 + NT * CHP * 2];
    __shared__ __align__(64) unsigned char smB[NT * CHP * 2];

    const int b     = blockIdx.y;
    const int sbase = blockIdx.x * NT;
    const int tid   = threadIdx.x;
    const int wave  = tid >> 5;
    const int lane  = tid & 31;
    const int l15   = lane & 15;
    const int hi    = lane >> 4;

    // ---- 1. gather lift tile: sLift[n][k] = feat[b, k, idx[n]]  (bf16) ----
    {
        const int n  = tid >> 2;            // 0..63
        const int kl = tid & 3;             // 0..3
        const int c  = idx[(size_t)b * NS + sbase + n];
        const float* src = feat + (size_t)b * FD * NC + c;
        __bf16* drow = (__bf16*)smA + (size_t)n * FDP;
        #pragma unroll 8
        for (int k0 = kl * 4; k0 < FD; k0 += 16) {
            const float f0 = src[(size_t)(k0 + 0) * NC];
            const float f1 = src[(size_t)(k0 + 1) * NC];
            const float f2 = src[(size_t)(k0 + 2) * NC];
            const float f3 = src[(size_t)(k0 + 3) * NC];
            drow[k0 + 0] = (__bf16)f0;
            drow[k0 + 1] = (__bf16)f1;
            drow[k0 + 2] = (__bf16)f2;
            drow[k0 + 3] = (__bf16)f3;
        }
    }
    __syncthreads();

    const v8f vz = {};
    v8f acc[2][4];

    // ---- 2. GEMM1: x = W_proj(256x768) @ lift ----
    #pragma unroll
    for (int mt = 0; mt < 2; ++mt)
        #pragma unroll
        for (int nt = 0; nt < 4; ++nt) acc[mt][nt] = vz;
    {
        const __bf16* sLift = (const __bf16*)smA;
        for (int k0 = 0; k0 < FD; k0 += 32) {
            v16bf bfr[4];
            #pragma unroll
            for (int nt = 0; nt < 4; ++nt)
                bfr[nt] = load_b_frag(sLift + (size_t)(nt * 16 + l15) * FDP, k0, hi);
            #pragma unroll
            for (int mt = 0; mt < 2; ++mt) {
                const __bf16* row = Wp + (size_t)((wave * 2 + mt) * 16 + l15) * FD;
                const v16bf af = load_a_frag(row, k0, hi);
                #pragma unroll
                for (int nt = 0; nt < 4; ++nt)
                    acc[mt][nt] = __builtin_amdgcn_wmma_f32_16x16x32_bf16(
                        false, af, false, bfr[nt], (short)0, acc[mt][nt],
                        false, false);
            }
        }
    }
    __syncthreads();   // all reads of sLift done: region A may be reused

    // ---- 3. epilogue 1: x + b_proj -> sXf (f32, aliases sLift) ----
    {
        float* sXf = (float*)smA;
        #pragma unroll
        for (int mt = 0; mt < 2; ++mt) {
            const int mbase = (wave * 2 + mt) * 16 + hi * 8;
            #pragma unroll
            for (int nt = 0; nt < 4; ++nt) {
                const int n = nt * 16 + l15;
                float* drow = sXf + (size_t)n * XFP + mbase;
                #pragma unroll
                for (int v = 0; v < 8; ++v)
                    drow[v] = acc[mt][nt][v] + bp[mbase + v];
            }
        }
    }
    __syncthreads();

    // ---- 4. LayerNorm over channels (256) per column ----
    const int nCol = tid & 63;   // 0..63
    const int seg  = tid >> 6;   // 0..3, each covers 64 channels
    {
        const float* sXf = (const float*)smA;
        const float* col = sXf + (size_t)nCol * XFP;
        float s1 = 0.f, s2 = 0.f;
        #pragma unroll
        for (int c = seg * 64; c < seg * 64 + 64; ++c) {
            const float x = col[c];
            s1 += x; s2 += x * x;
        }
        float* sRed = (float*)smB;
        sRed[seg * 64 + nCol]       = s1;
        sRed[256 + seg * 64 + nCol] = s2;
    }
    __syncthreads();
    {
        float* sRed = (float*)smB;
        float* sMu  = (float*)(smB + 2048);
        float* sRs  = sMu + 64;
        if (tid < 64) {
            float t1 = 0.f, t2 = 0.f;
            #pragma unroll
            for (int g = 0; g < 4; ++g) {
                t1 += sRed[g * 64 + tid];
                t2 += sRed[256 + g * 64 + tid];
            }
            const float mu  = t1 * (1.0f / CH);
            const float var = t2 * (1.0f / CH) - mu * mu;
            sMu[tid] = mu;
            sRs[tid] = rsqrtf(var + EPSV);
        }
    }
    __syncthreads();
    {
        const float* sXf = (const float*)smA;
        const float* sMu = (const float*)(smB + 2048);
        const float* sRs = sMu + 64;
        __bf16* sXb = (__bf16*)(smA + NT * XFP * 4);
        const float mu = sMu[nCol], rs = sRs[nCol];
        const float* col = sXf + (size_t)nCol * XFP;
        __bf16* orow = sXb + (size_t)nCol * CHP;
        #pragma unroll
        for (int c = seg * 64; c < seg * 64 + 64; ++c)
            orow[c] = (__bf16)((col[c] - mu) * rs);
    }
    __syncthreads();

    // ---- 5. GEMM2: h = relu(W_a1 @ x_ln + b_a1) ----
    #pragma unroll
    for (int mt = 0; mt < 2; ++mt)
        #pragma unroll
        for (int nt = 0; nt < 4; ++nt) acc[mt][nt] = vz;
    {
        const __bf16* sXb = (const __bf16*)(smA + NT * XFP * 4);
        for (int k0 = 0; k0 < CH; k0 += 32) {
            v16bf bfr[4];
            #pragma unroll
            for (int nt = 0; nt < 4; ++nt)
                bfr[nt] = load_b_frag(sXb + (size_t)(nt * 16 + l15) * CHP, k0, hi);
            #pragma unroll
            for (int mt = 0; mt < 2; ++mt) {
                const __bf16* row = Wa1 + (size_t)((wave * 2 + mt) * 16 + l15) * CH;
                const v16bf af = load_a_frag(row, k0, hi);
                #pragma unroll
                for (int nt = 0; nt < 4; ++nt)
                    acc[mt][nt] = __builtin_amdgcn_wmma_f32_16x16x32_bf16(
                        false, af, false, bfr[nt], (short)0, acc[mt][nt],
                        false, false);
            }
        }
    }
    // LN partials/mu are dead -> region B becomes sH
    {
        __bf16* sH = (__bf16*)smB;
        #pragma unroll
        for (int mt = 0; mt < 2; ++mt) {
            const int mbase = (wave * 2 + mt) * 16 + hi * 8;
            #pragma unroll
            for (int nt = 0; nt < 4; ++nt) {
                const int n = nt * 16 + l15;
                __bf16* drow = sH + (size_t)n * CHP + mbase;
                #pragma unroll
                for (int v = 0; v < 8; ++v) {
                    const float h = acc[mt][nt][v] + ba1[mbase + v];
                    drow[v] = (__bf16)(h > 0.f ? h : 0.f);
                }
            }
        }
    }
    __syncthreads();

    // ---- 6. GEMM3: f_lift = W_a2 @ h + b_a2 ; blend and store ----
    #pragma unroll
    for (int mt = 0; mt < 2; ++mt)
        #pragma unroll
        for (int nt = 0; nt < 4; ++nt) acc[mt][nt] = vz;
    {
        const __bf16* sH = (const __bf16*)smB;
        for (int k0 = 0; k0 < CH; k0 += 32) {
            v16bf bfr[4];
            #pragma unroll
            for (int nt = 0; nt < 4; ++nt)
                bfr[nt] = load_b_frag(sH + (size_t)(nt * 16 + l15) * CHP, k0, hi);
            #pragma unroll
            for (int mt = 0; mt < 2; ++mt) {
                const __bf16* row = Wa2 + (size_t)((wave * 2 + mt) * 16 + l15) * CH;
                const v16bf af = load_a_frag(row, k0, hi);
                #pragma unroll
                for (int nt = 0; nt < 4; ++nt)
                    acc[mt][nt] = __builtin_amdgcn_wmma_f32_16x16x32_bf16(
                        false, af, false, bfr[nt], (short)0, acc[mt][nt],
                        false, false);
            }
        }
    }
    #pragma unroll
    for (int mt = 0; mt < 2; ++mt) {
        const int mbase = (wave * 2 + mt) * 16 + hi * 8;
        #pragma unroll
        for (int nt = 0; nt < 4; ++nt) {
            const int n = nt * 16 + l15;
            const size_t base = ((size_t)b * CH + mbase) * NS + (sbase + n);
            #pragma unroll
            for (int v = 0; v < 8; ++v) {
                const float fl = acc[mt][nt][v] + ba2[mbase + v];
                const float g  = fgn[base + (size_t)v * NS];
                out[base + (size_t)v * NS] = (1.0f - ALPHA) * g + ALPHA * fl;
            }
        }
    }
}

// ---------------------------------------------------------------------------
extern "C" void kernel_launch(void* const* d_in, const int* in_sizes, int n_in,
                              void* d_out, int out_size, void* d_ws, size_t ws_size,
                              hipStream_t stream)
{
    (void)in_sizes; (void)n_in; (void)out_size; (void)ws_size;

    const float* seed = (const float*)d_in[0];
    const float* ctr  = (const float*)d_in[1];
    const float* feat = (const float*)d_in[2];
    const float* fgn  = (const float*)d_in[3];
    const float* Wp   = (const float*)d_in[4];
    const float* bp   = (const float*)d_in[5];
    const float* Wa1  = (const float*)d_in[6];
    const float* ba1  = (const float*)d_in[7];
    const float* Wa2  = (const float*)d_in[8];
    const float* ba2  = (const float*)d_in[9];
    float* out = (float*)d_out;

    char*   ws   = (char*)d_ws;
    int*    idx  = (int*)(ws + WS_IDX);
    __bf16* wpb  = (__bf16*)(ws + WS_WP);
    __bf16* wa1b = (__bf16*)(ws + WS_WA1);
    __bf16* wa2b = (__bf16*)(ws + WS_WA2);

    nn_argmin<<<dim3(NS / 256, BB), 256, 0, stream>>>(seed, ctr, idx);
    cvt_weights<<<(CH * FD + 255) / 256, 256, 0, stream>>>(Wp, Wa1, Wa2,
                                                           wpb, wa1b, wa2b);
    fused_lift<<<dim3(NS / NT, BB), 256, 0, stream>>>(feat, fgn, wpb, bp,
                                                      wa1b, ba1, wa2b, ba2,
                                                      idx, out);
}
